// CrossModalInteraction_57277683859938
// MI455X (gfx1250) — compile-verified, matrix operand order checked by reading, not run on hardware
//
#include <hip/hip_runtime.h>
#include <cstddef>
#include <cstdint>

// ---------------------------------------------------------------------------
// MI455X / gfx1250 implementation of the CrossModalInteraction block.
// wave32, WMMA f32<-f16 for all dense projections, lane-parallel selective
// scan with double-buffered ASYNC global->LDS staging of the dt/B/C rows.
// ---------------------------------------------------------------------------

typedef __attribute__((ext_vector_type(16))) _Float16 v16h;
typedef __attribute__((ext_vector_type(8)))  float    v8f;

#define HIDN   128
#define NST    8
#define KDIR   4
#define RNK    8
#define BB     4
#define HH     64
#define WW     64
#define LL     (HH * WW)      // 4096
#define ML     (BB * LL)      // 16384 rows per stream
#define CPROJ  24             // RANK + 2*NSTATE

// K-index inside a 16-bit 16x32 WMMA A/B fragment (ISA 7.12.2):
// lanes 0-15 hold K{0..7,16..23}, lanes 16-31 hold K{8..15,24..31}.
__device__ __forceinline__ int frag_k(int half, int j) {
    return (j < 8) ? (half * 8 + j) : (half * 8 + j + 8);
}

// Map scan position l of direction kdir to an index into xc_act[b][d][h*W+w].
__device__ __forceinline__ int xs_map(int kdir, int l) {
    int ll = (kdir >= 2) ? (LL - 1 - l) : l;
    if (kdir & 1) {                       // w-major (wh) ordering
        int w = ll / HH;
        int h = ll - w * HH;
        return h * WW + w;
    }
    return ll;                            // h-major (hw) ordering
}

__device__ __forceinline__ float silu_f(float x) {
    return x / (1.0f + __expf(-x));
}

// Async DMA one dword global -> LDS (tracked by ASYNCcnt). Generic shared
// pointers carry the LDS byte offset in addr[31:0] (ISA 10.2), so a 32-bit
// truncation yields the LDS address the instruction expects.
__device__ __forceinline__ void async_ld_dword(const float* lds_ptr, const float* gptr) {
    const unsigned lds_addr = (unsigned)(uintptr_t)lds_ptr;
    asm volatile("global_load_async_to_lds_b32 %0, %1, off"
                 :: "v"(lds_addr), "v"((unsigned long long)(uintptr_t)gptr)
                 : "memory");
}
__device__ __forceinline__ void wait_async0() {
    asm volatile("s_wait_asynccnt 0x0" ::: "memory");
}

// ---------------------------------------------------------------------------
// Generic WMMA GEMM: out[M,N] = A[M,K] * W[N,K]^T  (weights stored [outF,inF])
// block = 128 threads (4 waves). Each wave computes a 16(M) x 64(N) strip:
// one A fragment feeds 4 back-to-back WMMAs per k-step (4x A reuse).
// grid = (M/64, N/64).
// mode 0: plain store. mode 1: out = resid + gelu(acc + bias)   (exact erf)
// ---------------------------------------------------------------------------
__global__ void k_gemm_wmma(const float* __restrict__ A, const float* __restrict__ Wt,
                            const float* __restrict__ bias, const float* __restrict__ resid,
                            float* __restrict__ out, int M, int N, int Kd, int mode)
{
    const int lane = threadIdx.x & 31;
    const int wid  = threadIdx.x >> 5;        // 0..3 : m-tile within block
    const int half = lane >> 4;
    const int r    = lane & 15;
    const int mt   = blockIdx.x * 4 + wid;
    const int nt0  = blockIdx.y * 4;
    const int mrow = mt * 16 + r;

    v8f accs[4] = {};
    for (int k0 = 0; k0 < Kd; k0 += 32) {
        v16h a;
#pragma unroll
        for (int j = 0; j < 16; ++j)
            a[j] = (_Float16)A[(size_t)mrow * Kd + k0 + frag_k(half, j)];
#pragma unroll
        for (int t = 0; t < 4; ++t) {
            const int ncol = (nt0 + t) * 16 + r;
            v16h b;
#pragma unroll
            for (int j = 0; j < 16; ++j)
                b[j] = (_Float16)Wt[(size_t)ncol * Kd + k0 + frag_k(half, j)];
            accs[t] = __builtin_amdgcn_wmma_f32_16x16x32_f16(
                          false, a, false, b, (short)0, accs[t], false, false);
        }
    }

#pragma unroll
    for (int t = 0; t < 4; ++t) {
        const int n = (nt0 + t) * 16 + r;
#pragma unroll
        for (int i = 0; i < 8; ++i) {
            const int m = mt * 16 + i + 8 * half;
            float v = accs[t][i];
            if (mode == 1) {
                v += bias[n];
                v = 0.5f * v * (1.0f + erff(v * 0.70710678118f));   // exact GELU
                v += resid[(size_t)m * N + n];
            }
            out[(size_t)m * N + n] = v;
        }
    }
}

// ---------------------------------------------------------------------------
// Depthwise 3x3 conv (SAME, zero pad) + bias + SiLU.
// Input: xc half of xz, layout [s][b*L + l][256] (channels fastest -> coalesced
// reads). Output xc_act layout [s][b][d][l] (scan-friendly, contiguous in l).
// ---------------------------------------------------------------------------
__global__ void k_dwconv(const float* __restrict__ xz, const float* __restrict__ cw,
                         const float* __restrict__ cb, float* __restrict__ xc_act)
{
    size_t idx = (size_t)blockIdx.x * blockDim.x + threadIdx.x;
    if (idx >= (size_t)2 * BB * LL * HIDN) return;
    const int d = (int)(idx % HIDN);
    size_t p = idx / HIDN;
    const int l = (int)(p % LL);
    const int b = (int)((p / LL) % BB);
    const int s = (int)(p / ((size_t)LL * BB));
    const int hh = l / WW, ww2 = l % WW;

    const float* w9 = cw + ((size_t)s * HIDN + d) * 9;
    float acc = cb[s * HIDN + d];
#pragma unroll
    for (int dy = 0; dy < 3; ++dy) {
#pragma unroll
        for (int dx = 0; dx < 3; ++dx) {
            const int y = hh + dy - 1, x = ww2 + dx - 1;
            if (y >= 0 && y < HH && x >= 0 && x < WW) {
                acc += w9[dy * 3 + dx] *
                       xz[((size_t)s * ML + (size_t)b * LL + y * WW + x) * 256 + d];
            }
        }
    }
    xc_act[(((size_t)(s * BB + b)) * HIDN + d) * LL + l] = silu_f(acc);
}

// ---------------------------------------------------------------------------
// x_dbl[b,k,c,l] = sum_d xproj[s,k,c,d] * xs[b,k,d,l]  (c = 24, padded to 32)
// One wave per 16x16 (c,l) tile; the 4-direction permutation is applied on
// the B-fragment gather so xs is never materialized.
// grid = (L/16, 2, 2*B*KDIR), block = 32.
// ---------------------------------------------------------------------------
__global__ void k_xdbl_wmma(const float* __restrict__ xc_act,
                            const float* __restrict__ xproj,
                            float* __restrict__ xdbl)
{
    const int z    = blockIdx.z;
    const int kdir = z % KDIR;
    const int b    = (z / KDIR) % BB;
    const int s    = z / (KDIR * BB);
    const int lane = threadIdx.x;
    const int half = lane >> 4;
    const int r    = lane & 15;
    const int l    = blockIdx.x * 16 + r;
    const int ls   = xs_map(kdir, l);
    const int crow = blockIdx.y * 16 + r;

    const float* Ap = xproj  + ((size_t)(s * KDIR + kdir) * CPROJ) * HIDN;
    const float* Bp = xc_act + ((size_t)(s * BB + b) * HIDN) * LL;

    v8f acc = {};
    for (int k0 = 0; k0 < HIDN; k0 += 32) {
        v16h a, bb;
#pragma unroll
        for (int j = 0; j < 16; ++j) {
            const int kk = k0 + frag_k(half, j);
            a[j]  = (crow < CPROJ) ? (_Float16)Ap[(size_t)crow * HIDN + kk]
                                   : (_Float16)0.0f;
            bb[j] = (_Float16)Bp[(size_t)kk * LL + ls];
        }
        acc = __builtin_amdgcn_wmma_f32_16x16x32_f16(
                  false, a, false, bb, (short)0, acc, false, false);
    }

    float* outp = xdbl + (((size_t)(s * BB + b) * KDIR + kdir) * CPROJ) * LL;
    const int lcol = blockIdx.x * 16 + r;
#pragma unroll
    for (int i = 0; i < 8; ++i) {
        const int c = blockIdx.y * 16 + i + 8 * half;
        if (c < CPROJ) outp[(size_t)c * LL + lcol] = acc[i];
    }
}

// ---------------------------------------------------------------------------
// Selective scan. One block per (s,b,kdir) = 32 blocks x 1024 threads.
// tid = d*8 + n : each lane runs one independent scalar recurrence; the 8
// state lanes of a channel are contiguous so the y reduction is 3 shfl_xor
// steps inside a wave32. The shared dt/B/C chunk rows are DMA'd with
// GLOBAL_LOAD_ASYNC_TO_LDS_B32 into a double buffer (24 KB LDS) so the next
// chunk's transfer overlaps this chunk's recurrence. dt_proj+softplus fused.
// ---------------------------------------------------------------------------
__global__ void k_scan(const float* __restrict__ xdbl, const float* __restrict__ xc_act,
                       const float* __restrict__ dtw_all, const float* __restrict__ dtb_all,
                       const float* __restrict__ alogs, const float* __restrict__ ds_all,
                       float* __restrict__ out_y)
{
    const int blk  = blockIdx.x;
    const int kdir = blk % KDIR;
    const int b    = (blk / KDIR) % BB;
    const int s    = blk / (KDIR * BB);
    const int tid  = threadIdx.x;
    const int d    = tid >> 3;
    const int n    = tid & 7;

    __shared__ float s_dt[2][RNK][128];
    __shared__ float s_B[2][NST][128];
    __shared__ float s_C[2][NST][128];

    const float* xd   = xdbl + (((size_t)(s * BB + b) * KDIR + kdir) * CPROJ) * LL;
    const float* xrow = xc_act + ((size_t)(s * BB + b) * HIDN + d) * LL;

    float dtw[RNK];
#pragma unroll
    for (int rr = 0; rr < RNK; ++rr)
        dtw[rr] = dtw_all[(((size_t)(s * KDIR + kdir) * HIDN) + d) * RNK + rr];
    const float dtb = dtb_all[(size_t)(s * KDIR + kdir) * HIDN + d];
    const float An  = -__expf(alogs[((size_t)s * KDIR * HIDN + kdir * HIDN + d) * NST + n]);
    const float Dsv = ds_all[(size_t)s * KDIR * HIDN + kdir * HIDN + d];

    float h = 0.0f;
    float* oy = out_y + ((size_t)(s * BB + b) * KDIR + kdir) * LL * HIDN;

    const int rr = tid >> 7;      // 0..7 : chunk row this thread stages
    const int tc = tid & 127;     // position within chunk

    auto stage = [&](int p, int t0) {
        async_ld_dword(&s_dt[p][rr][tc], xd + (size_t)rr * LL + t0 + tc);
        async_ld_dword(&s_B[p][rr][tc],  xd + (size_t)(8 + rr) * LL + t0 + tc);
        async_ld_dword(&s_C[p][rr][tc],  xd + (size_t)(16 + rr) * LL + t0 + tc);
    };

    stage(0, 0);
    int p = 0;
    for (int t0 = 0; t0 < LL; t0 += 128, p ^= 1) {
        wait_async0();            // this thread's DMAs for buffer p done
        __syncthreads();          // ...and everyone else's
        if (t0 + 128 < LL) stage(p ^ 1, t0 + 128);   // overlap next chunk
        for (int tt = 0; tt < 128; ++tt) {
            const int t  = t0 + tt;
            const float x_t = xrow[xs_map(kdir, t)];
            float u = dtb;
#pragma unroll
            for (int q = 0; q < RNK; ++q) u += dtw[q] * s_dt[p][q][tt];
            const float dt = (u > 20.0f) ? u : log1pf(__expf(u));   // softplus
            h = h * __expf(dt * An) + dt * s_B[p][n][tt] * x_t;
            float val = h * s_C[p][n][tt];
            val += __shfl_xor(val, 1);
            val += __shfl_xor(val, 2);
            val += __shfl_xor(val, 4);
            if (n == 0) oy[(size_t)t * HIDN + d] = val + Dsv * x_t;
        }
    }
}

// ---------------------------------------------------------------------------
// Direction unwind + LayerNorm over D + SiLU(z) gate.
// One block per (s,b,l), 128 threads (d).
// ---------------------------------------------------------------------------
__global__ void k_merge_ln_gate(const float* __restrict__ out_y, const float* __restrict__ xz,
                                const float* __restrict__ ln_g, const float* __restrict__ ln_b,
                                float* __restrict__ ygated)
{
    __shared__ float red[HIDN];
    const int blk = blockIdx.x;
    const int l = blk % LL;
    const int b = (blk / LL) % BB;
    const int s = blk / (LL * BB);
    const int d = threadIdx.x;
    const int hh = l / WW, ww2 = l % WW;
    const int l_wh = ww2 * HH + hh;

    const float* oy = out_y + (size_t)((s * BB + b) * KDIR) * LL * HIDN;
    float v = oy[((size_t)0 * LL + l) * HIDN + d]
            + oy[((size_t)2 * LL + (LL - 1 - l)) * HIDN + d]
            + oy[((size_t)1 * LL + l_wh) * HIDN + d]
            + oy[((size_t)3 * LL + (LL - 1 - l_wh)) * HIDN + d];

    red[d] = v; __syncthreads();
    for (int off = HIDN / 2; off > 0; off >>= 1) { if (d < off) red[d] += red[d + off]; __syncthreads(); }
    const float mean = red[0] * (1.0f / HIDN);
    __syncthreads();
    const float dv = v - mean;
    red[d] = dv * dv; __syncthreads();
    for (int off = HIDN / 2; off > 0; off >>= 1) { if (d < off) red[d] += red[d + off]; __syncthreads(); }
    const float var = red[0] * (1.0f / HIDN);
    __syncthreads();

    const float nrm = dv * rsqrtf(var + 1e-5f) * ln_g[s * HIDN + d] + ln_b[s * HIDN + d];
    const float z   = xz[((size_t)s * ML + (size_t)b * LL + l) * 256 + HIDN + d];
    ygated[((size_t)s * ML + (size_t)b * LL + l) * HIDN + d] = nrm * silu_f(z);
}

// ---------------------------------------------------------------------------
// Cross-stream enhance LayerNorm: normed[s] = LN(guided[1-s]; enh_ln[s]).
// One block per (s, row), 128 threads.
// ---------------------------------------------------------------------------
__global__ void k_ln_enh(const float* __restrict__ guided, const float* __restrict__ g,
                         const float* __restrict__ bsh, float* __restrict__ normed)
{
    __shared__ float red[HIDN];
    const int blk = blockIdx.x;
    const int row = blk % ML;
    const int s   = blk / ML;
    const int d   = threadIdx.x;

    const float v = guided[(size_t)(1 - s) * ML * HIDN + (size_t)row * HIDN + d];
    red[d] = v; __syncthreads();
    for (int off = HIDN / 2; off > 0; off >>= 1) { if (d < off) red[d] += red[d + off]; __syncthreads(); }
    const float mean = red[0] * (1.0f / HIDN);
    __syncthreads();
    const float dv = v - mean;
    red[d] = dv * dv; __syncthreads();
    for (int off = HIDN / 2; off > 0; off >>= 1) { if (d < off) red[d] += red[d + off]; __syncthreads(); }
    const float var = red[0] * (1.0f / HIDN);
    __syncthreads();

    normed[((size_t)s * ML + row) * HIDN + d] =
        dv * rsqrtf(var + 1e-5f) * g[s * HIDN + d] + bsh[s * HIDN + d];
}

// ---------------------------------------------------------------------------
extern "C" void kernel_launch(void* const* d_in, const int* in_sizes, int n_in,
                              void* d_out, int out_size, void* d_ws, size_t ws_size,
                              hipStream_t stream)
{
    (void)in_sizes; (void)n_in; (void)out_size; (void)ws_size;

    const float* sar   = (const float*)d_in[0];
    const float* opt   = (const float*)d_in[1];
    const float* in_w  = (const float*)d_in[2];
    const float* cw    = (const float*)d_in[3];
    const float* cb    = (const float*)d_in[4];
    const float* xproj = (const float*)d_in[5];
    const float* dtw   = (const float*)d_in[6];
    const float* dtb   = (const float*)d_in[7];
    const float* alog  = (const float*)d_in[8];
    const float* dsv   = (const float*)d_in[9];
    const float* lng   = (const float*)d_in[10];
    const float* lnb   = (const float*)d_in[11];
    const float* outw  = (const float*)d_in[12];
    const float* elng  = (const float*)d_in[13];
    const float* elnb  = (const float*)d_in[14];
    const float* ew    = (const float*)d_in[15];
    const float* eb    = (const float*)d_in[16];
    float* out = (float*)d_out;

    float* ws     = (float*)d_ws;
    float* xz     = ws;                                              // 2*16384*256
    float* xc_act = xz     + (size_t)2 * ML * 256;                   // 2*4*128*4096
    float* xdbl   = xc_act + (size_t)2 * BB * HIDN * LL;             // 2*4*4*24*4096
    float* oy     = xdbl   + (size_t)2 * BB * KDIR * CPROJ * LL;     // 2*4*4*4096*128
    float* ygated = oy     + (size_t)2 * BB * KDIR * LL * HIDN;      // 2*16384*128
    float* guided = ygated + (size_t)2 * ML * HIDN;                  // 2*16384*128
    float* normed = guided + (size_t)2 * ML * HIDN;                  // 2*16384*128

    // 1) in_proj: xz = x @ in_w^T   (stream 0 = opt, stream 1 = sar)
    {
        dim3 g(ML / 64, 256 / 64);
        k_gemm_wmma<<<g, 128, 0, stream>>>(opt, in_w, nullptr, nullptr,
                                           xz, ML, 256, HIDN, 0);
        k_gemm_wmma<<<g, 128, 0, stream>>>(sar, in_w + (size_t)256 * HIDN, nullptr, nullptr,
                                           xz + (size_t)ML * 256, ML, 256, HIDN, 0);
    }
    // 2) depthwise conv 3x3 + bias + SiLU
    {
        size_t total = (size_t)2 * BB * LL * HIDN;
        k_dwconv<<<(unsigned)((total + 255) / 256), 256, 0, stream>>>(xz, cw, cb, xc_act);
    }
    // 3) x_dbl (WMMA, direction permutation fused into B gather)
    {
        dim3 g(LL / 16, 2, 2 * BB * KDIR);
        k_xdbl_wmma<<<g, 32, 0, stream>>>(xc_act, xproj, xdbl);
    }
    // 4) selective scan (dt_proj + softplus fused, async double-buffered LDS)
    k_scan<<<2 * BB * KDIR, 1024, 0, stream>>>(xdbl, xc_act, dtw, dtb, alog, dsv, oy);

    // 5) direction merge + LN + SiLU gate
    k_merge_ln_gate<<<2 * BB * LL, HIDN, 0, stream>>>(oy, xz, lng, lnb, ygated);

    // 6) out_proj (WMMA)
    {
        dim3 g(ML / 64, HIDN / 64);
        for (int s = 0; s < 2; ++s)
            k_gemm_wmma<<<g, 128, 0, stream>>>(ygated + (size_t)s * ML * HIDN,
                                               outw + (size_t)s * HIDN * HIDN,
                                               nullptr, nullptr,
                                               guided + (size_t)s * ML * HIDN,
                                               ML, HIDN, HIDN, 0);
    }
    // 7) enhance LN (cross-stream)
    k_ln_enh<<<2 * ML, HIDN, 0, stream>>>(guided, elng, elnb, normed);

    // 8) enhance GEMM + GELU + residual  ->  d_out (out0 then out1)
    {
        dim3 g(ML / 64, HIDN / 64);
        k_gemm_wmma<<<g, 128, 0, stream>>>(normed, ew, eb, sar,
                                           out, ML, HIDN, HIDN, 1);
        k_gemm_wmma<<<g, 128, 0, stream>>>(normed + (size_t)ML * HIDN,
                                           ew + (size_t)HIDN * HIDN, eb + HIDN, opt,
                                           out + (size_t)ML * HIDN, ML, HIDN, HIDN, 1);
    }
}
